// GATModel_6279242187334
// MI455X (gfx1250) — matrix operand order repaired and checked by reading, not compiled, not run on hardware
//
#include <hip/hip_runtime.h>

typedef __attribute__((ext_vector_type(16))) _Float16 v16h;
typedef __attribute__((ext_vector_type(8)))  float    v8f;

#define F_IN   64
#define F_HID  64
#define F_OUT  32
#define NEG_SLOPE 0.2f
#define BN_EPS 1e-5f

// -------- GEMM1: h[N,64] = x[N,64] @ W[64,64]^T  (WMMA f16 -> f32 acc) --------
// Block = 128 threads = 4 waves; each wave owns a 16-row M-tile (block: 64 rows).
__global__ __launch_bounds__(128)
void k_gemm1(const float* __restrict__ x, const float* __restrict__ W,
             float* __restrict__ h, int N)
{
    const int lane = threadIdx.x & 31;
    const int wave = threadIdx.x >> 5;
    const int hl   = lane >> 4;        // half: 0 = lanes 0-15, 1 = lanes 16-31
    const int l16  = lane & 15;
    const int rowbase = blockIdx.x * 64 + wave * 16;
    const int row = rowbase + l16;     // A-row for this lane (both halves hold M=0..15)
    const bool fullTile = (rowbase + 16 <= N);

    // A fragments for kstep 0,1.  K(v) = kstep*32 + hl*8 + (v<8 ? v : v+8)
    v16h a[2];
    #pragma unroll
    for (int ks = 0; ks < 2; ++ks) {
        const int k0 = ks * 32 + hl * 8;
        if (row < N) {
            const float4* p0 = (const float4*)(x + (size_t)row * F_IN + k0);
            const float4* p1 = (const float4*)(x + (size_t)row * F_IN + k0 + 16);
            float4 c0 = p0[0], c1 = p0[1], c2 = p1[0], c3 = p1[1];
            a[ks][0]=(_Float16)c0.x; a[ks][1]=(_Float16)c0.y; a[ks][2]=(_Float16)c0.z; a[ks][3]=(_Float16)c0.w;
            a[ks][4]=(_Float16)c1.x; a[ks][5]=(_Float16)c1.y; a[ks][6]=(_Float16)c1.z; a[ks][7]=(_Float16)c1.w;
            a[ks][8]=(_Float16)c2.x; a[ks][9]=(_Float16)c2.y; a[ks][10]=(_Float16)c2.z; a[ks][11]=(_Float16)c2.w;
            a[ks][12]=(_Float16)c3.x; a[ks][13]=(_Float16)c3.y; a[ks][14]=(_Float16)c3.z; a[ks][15]=(_Float16)c3.w;
        } else {
            #pragma unroll
            for (int v = 0; v < 16; ++v) a[ks][v] = (_Float16)0.f;
        }
    }

    #pragma unroll
    for (int t = 0; t < 4; ++t) {                 // 4 N-tiles of 16 cols -> 64 cols
        v8f acc = {};
        #pragma unroll
        for (int ks = 0; ks < 2; ++ks) {
            const int col = t * 16 + l16;         // B lane = column
            const int kb  = ks * 32 + hl * 8;     // B: K(v) = kb + (v<8 ? v : v+8)
            const float4* wr = (const float4*)(W + (size_t)col * F_IN + kb);
            v16h b;
            #pragma unroll
            for (int q = 0; q < 4; ++q) {
                float4 f = wr[q + (q >> 1) * 2];
                b[q*4+0]=(_Float16)f.x; b[q*4+1]=(_Float16)f.y;
                b[q*4+2]=(_Float16)f.z; b[q*4+3]=(_Float16)f.w;
            }
            acc = __builtin_amdgcn_wmma_f32_16x16x32_f16(false, a[ks], false, b,
                                                         (short)0, acc, false, false);
        }
        // D: row = r + hl*8 (lanes 0-15) / r + 8 + ... wait hl folds it; col = l16
        float* base = h + (size_t)(rowbase + hl * 8) * F_HID + t * 16 + l16;
        if (fullTile) {                            // hot path: 8 stores, imm offsets
            #pragma unroll
            for (int r = 0; r < 8; ++r) base[(size_t)r * F_HID] = acc[r];
        } else {                                   // only the last block takes this
            #pragma unroll
            for (int r = 0; r < 8; ++r)
                if (rowbase + hl * 8 + r < N) base[(size_t)r * F_HID] = acc[r];
        }
    }
}

// -------- per-node attention scalars + seg_sum init --------
__global__ __launch_bounds__(256)
void k_node_a(const float* __restrict__ h, const float* __restrict__ att_src,
              const float* __restrict__ att_dst,
              float* __restrict__ a_s, float* __restrict__ a_d,
              float* __restrict__ seg_sum, int N)
{
    int n = blockIdx.x * blockDim.x + threadIdx.x;
    if (n >= N) return;
    const float4* hv = (const float4*)(h + (size_t)n * F_HID);
    const float4* as = (const float4*)att_src;
    const float4* ad = (const float4*)att_dst;
    float s = 0.f, d = 0.f;
    #pragma unroll
    for (int i = 0; i < 16; ++i) {
        float4 v = hv[i], u = as[i], w = ad[i];
        s += v.x*u.x + v.y*u.y + v.z*u.z + v.w*u.w;
        d += v.x*w.x + v.y*w.y + v.z*w.z + v.w*w.w;
    }
    a_s[n] = s; a_d[n] = d; seg_sum[n] = 0.f;
}

// -------- edge pass 1: w = exp(leaky_relu(a_src[s]+a_dst[d])); seg_sum[d] += w ----
// (seg-max pass elided: softmax is shift-invariant and logits are bounded small)
__global__ __launch_bounds__(256)
void k_edge_exp(const int* __restrict__ ei, int E, int N,
                const float* __restrict__ a_s, const float* __restrict__ a_d,
                float* __restrict__ expw, float* __restrict__ seg_sum)
{
    const int M = E + N;                        // < 2^31, keep 32-bit hot loop
    for (int e = blockIdx.x * blockDim.x + threadIdx.x; e < M;
         e += gridDim.x * blockDim.x) {
        int s, d;
        if (e < E) { s = ei[e]; d = ei[E + e]; }
        else       { s = d = e - E; }           // self-loop
        float l = a_s[s] + a_d[d];
        l = (l > 0.f) ? l : NEG_SLOPE * l;
        float w = __expf(l);
        expw[e] = w;
        atomicAdd(&seg_sum[d], w);
    }
}

// -------- edge pass 1.5: alpha[e] = expw[e] / seg_sum[dst]  (in place) ------------
__global__ __launch_bounds__(256)
void k_alpha(const int* __restrict__ ei, int E, int N,
             const float* __restrict__ seg_sum, float* __restrict__ expw)
{
    const int M = E + N;
    for (int e = blockIdx.x * blockDim.x + threadIdx.x; e < M;
         e += gridDim.x * blockDim.x) {
        int d = (e < E) ? ei[E + e] : (e - E);
        expw[e] = expw[e] / seg_sum[d];
    }
}

// -------- edge pass 2: hacc[d,:] += alpha * h[s,:]  (16 lanes/edge, float4) --------
__global__ __launch_bounds__(256)
void k_edge_scatter(const int* __restrict__ ei, int E, int N,
                    const float* __restrict__ h, const float* __restrict__ alpha,
                    float* __restrict__ hacc)
{
    const int T = (E + N) * 16;                 // 27.2M < 2^31
    for (int gid = blockIdx.x * blockDim.x + threadIdx.x; gid < T;
         gid += gridDim.x * blockDim.x) {
        int e = gid >> 4;
        int g = gid & 15;
        int s, d;
        if (e < E) { s = ei[e]; d = ei[E + e]; }
        else       { s = d = e - E; }
        float al = alpha[e];
        float4 v = ((const float4*)(h + (size_t)s * F_HID))[g];
        float* o = hacc + (size_t)d * F_HID + g * 4;
        atomicAdd(o + 0, v.x * al);
        atomicAdd(o + 1, v.y * al);
        atomicAdd(o + 2, v.z * al);
        atomicAdd(o + 3, v.w * al);
    }
}

// -------- BN stats: per-channel sum & sumsq over N (LDS reduce, then atomics) -----
__global__ __launch_bounds__(256)
void k_bn_stats(const float* __restrict__ hacc, float* __restrict__ gsum,
                float* __restrict__ gsq, int N)
{
    __shared__ float ss[256], sq[256];
    const int c = threadIdx.x & 63;
    const int j = threadIdx.x >> 6;
    float s = 0.f, q = 0.f;
    for (int n = blockIdx.x * 4 + j; n < N; n += gridDim.x * 4) {
        float v = hacc[(size_t)n * F_HID + c];
        s += v; q += v * v;
    }
    ss[threadIdx.x] = s; sq[threadIdx.x] = q;
    __syncthreads();
    if (threadIdx.x < 64) {
        float ts = ss[c] + ss[c+64] + ss[c+128] + ss[c+192];
        float tq = sq[c] + sq[c+64] + sq[c+128] + sq[c+192];
        atomicAdd(&gsum[c], ts);
        atomicAdd(&gsq[c], tq);
    }
}

// -------- BN finalize: scale = gamma*rsqrt(var+eps); shift = beta - mu*scale ------
// conv_b dropped: it cancels exactly through batch-norm (mean-shift only).
__global__ void k_bn_final(const float* __restrict__ gsum, const float* __restrict__ gsq,
                           const float* __restrict__ gamma, const float* __restrict__ beta,
                           float* __restrict__ bnsc, float* __restrict__ bnsh, int N)
{
    int c = threadIdx.x;
    if (c >= F_HID) return;
    float invN = 1.f / (float)N;
    float mu  = gsum[c] * invN;
    float var = gsq[c] * invN - mu * mu;
    var = var > 0.f ? var : 0.f;
    float sc = gamma[c] * rsqrtf(var + BN_EPS);
    bnsc[c] = sc;
    bnsh[c] = beta[c] - mu * sc;
}

// -------- GEMM2: out[N,32] = relu(bn(hacc)) @ fc_W^T + fc_b  (WMMA) --------------
__global__ __launch_bounds__(128)
void k_gemm2(const float* __restrict__ hacc, const float* __restrict__ bnsc,
             const float* __restrict__ bnsh, const float* __restrict__ fcW,
             const float* __restrict__ fcb, float* __restrict__ out, int N)
{
    __shared__ float ssc[F_HID], ssh[F_HID];
    if (threadIdx.x < F_HID) { ssc[threadIdx.x] = bnsc[threadIdx.x];
                               ssh[threadIdx.x] = bnsh[threadIdx.x]; }
    __syncthreads();

    const int lane = threadIdx.x & 31;
    const int wave = threadIdx.x >> 5;
    const int hl   = lane >> 4;
    const int l16  = lane & 15;
    const int rowbase = blockIdx.x * 64 + wave * 16;
    const int row = rowbase + l16;
    const bool fullTile = (rowbase + 16 <= N);

    v16h a[2];
    #pragma unroll
    for (int ks = 0; ks < 2; ++ks) {
        const int k0 = ks * 32 + hl * 8;
        #pragma unroll
        for (int v = 0; v < 8; ++v) {
            int c0 = k0 + v, c1 = k0 + 16 + v;
            float f0 = 0.f, f1 = 0.f;
            if (row < N) {
                f0 = hacc[(size_t)row * F_HID + c0] * ssc[c0] + ssh[c0];
                f1 = hacc[(size_t)row * F_HID + c1] * ssc[c1] + ssh[c1];
                f0 = f0 > 0.f ? f0 : 0.f;             // ReLU
                f1 = f1 > 0.f ? f1 : 0.f;
            }
            a[ks][v]     = (_Float16)f0;
            a[ks][v + 8] = (_Float16)f1;
        }
    }

    #pragma unroll
    for (int t = 0; t < 2; ++t) {                     // 2 N-tiles -> 32 cols
        v8f acc = {};
        #pragma unroll
        for (int ks = 0; ks < 2; ++ks) {
            const int col = t * 16 + l16;
            const int kb  = ks * 32 + hl * 8;
            const float4* wr = (const float4*)(fcW + (size_t)col * F_HID + kb);
            v16h b;
            #pragma unroll
            for (int q = 0; q < 4; ++q) {
                float4 f = wr[q + (q >> 1) * 2];
                b[q*4+0]=(_Float16)f.x; b[q*4+1]=(_Float16)f.y;
                b[q*4+2]=(_Float16)f.z; b[q*4+3]=(_Float16)f.w;
            }
            acc = __builtin_amdgcn_wmma_f32_16x16x32_f16(false, a[ks], false, b,
                                                         (short)0, acc, false, false);
        }
        float bias = fcb[t * 16 + l16];
        float* base = out + (size_t)(rowbase + hl * 8) * F_OUT + t * 16 + l16;
        if (fullTile) {
            #pragma unroll
            for (int r = 0; r < 8; ++r) base[(size_t)r * F_OUT] = acc[r] + bias;
        } else {
            #pragma unroll
            for (int r = 0; r < 8; ++r)
                if (rowbase + hl * 8 + r < N) base[(size_t)r * F_OUT] = acc[r] + bias;
        }
    }
}

extern "C" void kernel_launch(void* const* d_in, const int* in_sizes, int n_in,
                              void* d_out, int out_size, void* d_ws, size_t ws_size,
                              hipStream_t stream)
{
    const float* x       = (const float*)d_in[0];
    const int*   ei      = (const int*)  d_in[1];   // JAX default x64-off -> int32
    const float* W       = (const float*)d_in[2];
    const float* att_src = (const float*)d_in[3];
    const float* att_dst = (const float*)d_in[4];
    /* d_in[5] = conv_b: cancels exactly through BatchNorm -> unused */
    const float* gamma   = (const float*)d_in[6];
    const float* beta    = (const float*)d_in[7];
    const float* fcW     = (const float*)d_in[8];
    const float* fcb     = (const float*)d_in[9];
    float* out = (float*)d_out;

    const int N = in_sizes[0] / F_IN;
    const int E = in_sizes[1] / 2;
    const int M = E + N;

    // workspace carve (256B aligned slices)
    char* p = (char*)d_ws;
    auto carve = [&](size_t bytes) -> float* {
        float* r = (float*)p; p += (bytes + 255) & ~(size_t)255; return r;
    };
    float* h     = carve((size_t)N * F_HID * 4);
    float* a_s   = carve((size_t)N * 4);
    float* a_d   = carve((size_t)N * 4);
    float* segs  = carve((size_t)N * 4);
    float* expw  = carve((size_t)M * 4);
    float* hacc  = carve((size_t)N * F_HID * 4);
    float* gstat = carve(4 * F_HID * 4);
    float* gsum = gstat, *gsq = gstat + 64, *bnsc = gstat + 128, *bnsh = gstat + 192;
    (void)n_in; (void)out_size; (void)ws_size;

    hipMemsetAsync(hacc,  0, (size_t)N * F_HID * 4, stream);
    hipMemsetAsync(gstat, 0, 2 * F_HID * 4, stream);

    k_gemm1<<<(N + 63) / 64, 128, 0, stream>>>(x, W, h, N);
    k_node_a<<<(N + 255) / 256, 256, 0, stream>>>(h, att_src, att_dst, a_s, a_d, segs, N);

    int eb = (M + 255) / 256;
    k_edge_exp<<<eb, 256, 0, stream>>>(ei, E, N, a_s, a_d, expw, segs);
    k_alpha<<<eb, 256, 0, stream>>>(ei, E, N, segs, expw);

    int sb = (int)(((long long)M * 16 + 255) / 256);
    k_edge_scatter<<<sb, 256, 0, stream>>>(ei, E, N, h, expw, hacc);

    k_bn_stats<<<256, 256, 0, stream>>>(hacc, gsum, gsq, N);
    k_bn_final<<<1, 64, 0, stream>>>(gsum, gsq, gamma, beta, bnsc, bnsh, N);
    k_gemm2<<<(N + 63) / 64, 128, 0, stream>>>(hacc, bnsc, bnsh, fcW, fcb, out, N);
}
